// CGCN_27986006901064
// MI455X (gfx1250) — matrix operand order repaired and checked by reading, not compiled, not run on hardware
//
#include <hip/hip_runtime.h>
#include <hip/hip_bf16.h>
#include <math.h>

// ---------------- CDNA5 WMMA types ----------------
typedef __attribute__((ext_vector_type(16))) __bf16 v16bf;
typedef __attribute__((ext_vector_type(8)))  __bf16 v8bf;
typedef __attribute__((ext_vector_type(8)))  float  v8f;

#define WMMA_BF16(a, b, c) \
  __builtin_amdgcn_wmma_f32_16x16x32_bf16(false, (a), false, (b), (short)0, (c), false, false)

__device__ __forceinline__ unsigned short f32_to_bf16_rne(float f) {
  unsigned int u = __float_as_uint(f);
  unsigned int r = u + 0x7fffu + ((u >> 16) & 1u);
  return (unsigned short)(r >> 16);
}

// A-matrix fragment (16x32 bf16, row-major source, ld in elements).
// ISA 7.12.2: lanes 0-15 row M=lane, VGPR0..3 = K{0..7}, VGPR4..7 = K{16..23};
// lanes 16-31 same rows, K offset +8.  => two contiguous 16B loads per lane.
__device__ __forceinline__ v16bf load_fragA(const unsigned short* base, int ld,
                                            int row0, int k0, int lane) {
  int lr = lane & 15, half = lane >> 4;
  const unsigned short* p = base + (size_t)(row0 + lr) * ld + k0 + half * 8;
  v8bf lo = *reinterpret_cast<const v8bf*>(p);
  v8bf hi = *reinterpret_cast<const v8bf*>(p + 16);
  return __builtin_shufflevector(lo, hi, 0,1,2,3,4,5,6,7,8,9,10,11,12,13,14,15);
}

// B-matrix fragment (32x16 bf16) for NT gemm: BT stored row-major [Nout,K].
// lanes 0-15 col N=lane hold K=k0..k0+7 in VGPR0..3, K=k0+16..k0+23 in VGPR4..7;
// lanes 16-31 same cols, K offset +8.
__device__ __forceinline__ v16bf load_fragB(const unsigned short* base, int ld,
                                            int col0, int k0, int lane) {
  int lc = lane & 15, half = lane >> 4;
  const unsigned short* p = base + (size_t)(col0 + lc) * ld + k0 + half * 8;
  v8bf lo = *reinterpret_cast<const v8bf*>(p);
  v8bf hi = *reinterpret_cast<const v8bf*>(p + 16);
  return __builtin_shufflevector(lo, hi, 0,1,2,3,4,5,6,7,8,9,10,11,12,13,14,15);
}

__device__ __forceinline__ float wave_sum(float v) {
#pragma unroll
  for (int off = 16; off > 0; off >>= 1) v += __shfl_xor(v, off, 32);
  return v;
}

// ---------------- elementwise / prep kernels ----------------
__global__ void k_zero(float* p, size_t n) {
  size_t i = (size_t)blockIdx.x * blockDim.x + threadIdx.x;
  if (i < n) p[i] = 0.f;
}

__global__ void k_cvt_bf16(const float* s, unsigned short* d, size_t n) {
  size_t i = (size_t)blockIdx.x * blockDim.x + threadIdx.x;
  if (i < n) d[i] = f32_to_bf16_rne(s[i]);
}

// W [K,Nout] f32 -> WT [Nout,K] bf16
__global__ void k_wt(const float* W, unsigned short* WT, int K, int Nout) {
  size_t i = (size_t)blockIdx.x * blockDim.x + threadIdx.x;
  if (i >= (size_t)K * Nout) return;
  int k = (int)(i / Nout), n = (int)(i % Nout);
  WT[(size_t)n * K + k] = f32_to_bf16_rne(W[i]);
}

// ---------------- 16-wide NT WMMA GEMM (for Nout == 16) ----------------
__global__ __launch_bounds__(256)
void k_gemm_nt(const unsigned short* __restrict__ A,
               const unsigned short* __restrict__ BT,
               const float* __restrict__ bias, float* __restrict__ C,
               int M, int Nout, int K, float alpha) {
  int lane = threadIdx.x & 31;
  int wave = threadIdx.x >> 5;
  int tilesN = Nout >> 4;
  int nTiles = (M >> 4) * tilesN;
  int tile = blockIdx.x * 8 + wave;
  if (tile >= nTiles) return;
  int tm = tile / tilesN, tn = tile % tilesN;

  v8f c = {0.f, 0.f, 0.f, 0.f, 0.f, 0.f, 0.f, 0.f};
  for (int k0 = 0; k0 < K; k0 += 32) {
    v16bf a = load_fragA(A, K, tm * 16, k0, lane);
    v16bf b = load_fragB(BT, K, tn * 16, k0, lane);
    c = WMMA_BF16(a, b, c);
  }
  int n = tn * 16 + (lane & 15);
  int mbase = tm * 16 + ((lane >> 4) << 3);
  float bv = bias ? bias[n] : 0.f;
#pragma unroll
  for (int r = 0; r < 8; ++r)
    C[(size_t)(mbase + r) * Nout + n] = c[r] * alpha + bv;
}

// ---------------- 2x4 register-blocked NT WMMA GEMM ----------------
// Wave computes a 32x64 output macro-tile (8 accumulators); per k-step:
// 12 b128 loads feed 8 v_wmma (1.5:1).  Requires M%32==0 && Nout%64==0.
__global__ __launch_bounds__(256)
void k_gemm_nt24(const unsigned short* __restrict__ A,
                 const unsigned short* __restrict__ BT,
                 const float* __restrict__ bias, float* __restrict__ C,
                 int M, int Nout, int K, float alpha) {
  int lane = threadIdx.x & 31;
  int wave = threadIdx.x >> 5;
  int tilesN = Nout >> 6;
  int nTiles = (M >> 5) * tilesN;
  int tile = blockIdx.x * 8 + wave;
  if (tile >= nTiles) return;
  int tm = (tile / tilesN) * 32, tn = (tile % tilesN) * 64;

  v8f acc[2][4];
#pragma unroll
  for (int i = 0; i < 2; ++i)
#pragma unroll
    for (int j = 0; j < 4; ++j)
      acc[i][j] = (v8f){0.f,0.f,0.f,0.f,0.f,0.f,0.f,0.f};

  for (int k0 = 0; k0 < K; k0 += 32) {
    v16bf af[2], bf[4];
#pragma unroll
    for (int i = 0; i < 2; ++i) af[i] = load_fragA(A, K, tm + i * 16, k0, lane);
#pragma unroll
    for (int j = 0; j < 4; ++j) bf[j] = load_fragB(BT, K, tn + j * 16, k0, lane);
#pragma unroll
    for (int i = 0; i < 2; ++i)
#pragma unroll
      for (int j = 0; j < 4; ++j)
        acc[i][j] = WMMA_BF16(af[i], bf[j], acc[i][j]);
  }

  int nc = lane & 15;
  int mb = (lane >> 4) << 3;
#pragma unroll
  for (int j = 0; j < 4; ++j) {
    float bv = bias ? bias[tn + j * 16 + nc] : 0.f;
#pragma unroll
    for (int i = 0; i < 2; ++i)
#pragma unroll
      for (int r = 0; r < 8; ++r)
        C[(size_t)(tm + i * 16 + mb + r) * Nout + tn + j * 16 + nc] =
            acc[i][j][r] * alpha + bv;
  }
}

// ---------------- Barlow-Twins fused WMMA reduction (4x4 blocked) ----------------
// bt = Z1n @ Z2n^T [N,N]; accumulate sum(bt^2), sum((diag-1)^2), sum(diag^2)
// Wave computes a 64x64 macro-tile: per k-step 16 b128 loads feed 16 v_wmma (1:1).
// No store epilogue -> register headroom for the deep accumulator file.
__global__ __launch_bounds__(256)
void k_bt_reduce44(const unsigned short* __restrict__ Z1,
                   const unsigned short* __restrict__ Z2,
                   int N, int K, float* bucket_sumsq,
                   float* acc_diag_err, float* acc_diag_sq) {
  int lane = threadIdx.x & 31;
  int wave = threadIdx.x >> 5;
  int tilesN = N >> 6;
  int tile = blockIdx.x * 8 + wave;
  int tm4 = tile / tilesN, tn4 = tile % tilesN;
  int tm = tm4 * 64, tn = tn4 * 64;

  v8f acc[4][4];
#pragma unroll
  for (int i = 0; i < 4; ++i)
#pragma unroll
    for (int j = 0; j < 4; ++j)
      acc[i][j] = (v8f){0.f,0.f,0.f,0.f,0.f,0.f,0.f,0.f};

  for (int k0 = 0; k0 < K; k0 += 32) {
    v16bf af[4], bf[4];
#pragma unroll
    for (int i = 0; i < 4; ++i) af[i] = load_fragA(Z1, K, tm + i * 16, k0, lane);
#pragma unroll
    for (int j = 0; j < 4; ++j) bf[j] = load_fragB(Z2, K, tn + j * 16, k0, lane);
#pragma unroll
    for (int i = 0; i < 4; ++i)
#pragma unroll
      for (int j = 0; j < 4; ++j)
        acc[i][j] = WMMA_BF16(af[i], bf[j], acc[i][j]);
  }

  float ss = 0.f;
#pragma unroll
  for (int i = 0; i < 4; ++i)
#pragma unroll
    for (int j = 0; j < 4; ++j)
#pragma unroll
      for (int r = 0; r < 8; ++r)
        ss += acc[i][j][r] * acc[i][j][r];
  ss = wave_sum(ss);
  if (lane == 0) atomicAdd(&bucket_sumsq[tile & 2047], ss);

  if (tm4 == tn4) {
    // diagonal lives in sub-tiles acc[i][i]; element (M = r + 8*(lane>=16),
    // Ncol = lane&15) is diagonal when lane==r (lanes 0..7) or lane==r+24.
    float de = 0.f, ds = 0.f;
#pragma unroll
    for (int i = 0; i < 4; ++i) {
      float d = 0.f, valid = 0.f;
#pragma unroll
      for (int r = 0; r < 8; ++r) {
        bool sel = (lane == r) || (lane == r + 24);
        d += sel ? acc[i][i][r] : 0.f;
        valid += sel ? 1.f : 0.f;
      }
      de += valid * (d - 1.f) * (d - 1.f);
      ds += valid * d * d;
    }
    de = wave_sum(de);
    ds = wave_sum(ds);
    if (lane == 0) {
      atomicAdd(acc_diag_err, de);
      atomicAdd(acc_diag_sq, ds);
    }
  }
}

// ---------------- SPMM (atomic scatter-add) ----------------
__global__ void k_spmm(const float* __restrict__ val, const int* __restrict__ rows,
                       const int* __restrict__ cols, const float* __restrict__ S,
                       float* __restrict__ Hout, int E, int Hdim) {
  size_t tid = (size_t)blockIdx.x * blockDim.x + threadIdx.x;
  size_t e = tid >> 5;
  if (e >= (size_t)E) return;
  int seg = ((int)tid & 31) * 8;  // Hdim == 256 = 32 lanes * 8
  float v = val[e];
  int r = rows[e], cc = cols[e];
  const float* src = S + (size_t)cc * Hdim + seg;
  float* dst = Hout + (size_t)r * Hdim + seg;
#pragma unroll
  for (int j = 0; j < 8; ++j) atomicAdd(dst + j, v * src[j]);
}

__global__ void k_bias_relu(float* h, const float* b, size_t n, int Hdim) {
  size_t i = (size_t)blockIdx.x * blockDim.x + threadIdx.x;
  if (i < n) h[i] = fmaxf(h[i] + b[i % Hdim], 0.f);
}

__global__ void k_avg(const float* z1, const float* z2, float* zout,
                      unsigned short* zb, size_t n) {
  size_t i = (size_t)blockIdx.x * blockDim.x + threadIdx.x;
  if (i >= n) return;
  float v = 0.5f * (z1[i] + z2[i]);
  zout[i] = v;
  zb[i] = f32_to_bf16_rne(v);
}

// wave-per-row L2 normalize -> bf16
__global__ void k_rownorm(const float* __restrict__ Z, unsigned short* __restrict__ Zn,
                          int N, int Hdim) {
  int lane = threadIdx.x & 31;
  int row = blockIdx.x * 8 + (threadIdx.x >> 5);
  if (row >= N) return;
  const float* p = Z + (size_t)row * Hdim;
  float ss = 0.f;
  for (int j = lane; j < Hdim; j += 32) { float x = p[j]; ss += x * x; }
  ss = wave_sum(ss);
  float inv = 1.f / fmaxf(sqrtf(ss), 1e-12f);
  unsigned short* q = Zn + (size_t)row * Hdim;
  for (int j = lane; j < Hdim; j += 32) q[j] = f32_to_bf16_rne(p[j] * inv);
}

__global__ void k_logsoftmax(const float* __restrict__ logits, float* __restrict__ out,
                             int N, int C) {
  int i = blockIdx.x * blockDim.x + threadIdx.x;
  if (i >= N) return;
  const float* p = logits + (size_t)i * C;
  float m = -1e30f;
  for (int j = 0; j < C; ++j) m = fmaxf(m, p[j]);
  float s = 0.f;
  for (int j = 0; j < C; ++j) s += expf(p[j] - m);
  float l = logf(s) + m;
  float* q = out + (size_t)i * C;
  for (int j = 0; j < C; ++j) q[j] = p[j] - l;
}

__global__ void k_nll(const float* __restrict__ out, const int* __restrict__ train_id,
                      const int* __restrict__ y, float* acc, int T, int C) {
  int t = blockIdx.x * blockDim.x + threadIdx.x;
  if (t >= T) return;
  int id = train_id[t];
  int lab = y[id];
  atomicAdd(acc, -out[(size_t)id * C + lab]);
}

__global__ void k_gather_rows(const unsigned short* __restrict__ Zn,
                              const int* __restrict__ ids,
                              unsigned short* __restrict__ dst, int T, int Hdim) {
  size_t i = (size_t)blockIdx.x * blockDim.x + threadIdx.x;
  if (i >= (size_t)T * Hdim) return;
  int t = (int)(i / Hdim), j = (int)(i % Hdim);
  dst[i] = Zn[(size_t)ids[t] * Hdim + j];
}

__global__ void k_gather_lab(const int* y, const int* ids, int* lab, int T) {
  int t = blockIdx.x * blockDim.x + threadIdx.x;
  if (t < T) lab[t] = y[ids[t]];
}

// per-row logsumexp(all) - logsumexp(pos) with -1e9 masking, block per row
__global__ __launch_bounds__(256)
void k_supcon(const float* __restrict__ sim, const int* __restrict__ lab,
              float* acc, int T) {
  __shared__ float r1[256], r2[256];
  int i = blockIdx.x, t = threadIdx.x;
  int li = lab[i];
  const float* row = sim + (size_t)i * T;
  float m_all = -1e30f, m_pos = -1e30f;
  for (int j = t; j < T; j += 256) {
    float s = row[j];
    m_all = fmaxf(m_all, s);
    float sp = (lab[j] == li) ? s : -1e9f;
    m_pos = fmaxf(m_pos, sp);
  }
  r1[t] = m_all; r2[t] = m_pos;
  __syncthreads();
  for (int off = 128; off > 0; off >>= 1) {
    if (t < off) { r1[t] = fmaxf(r1[t], r1[t + off]); r2[t] = fmaxf(r2[t], r2[t + off]); }
    __syncthreads();
  }
  m_all = r1[0]; m_pos = r2[0];
  __syncthreads();
  float s_all = 0.f, s_pos = 0.f;
  for (int j = t; j < T; j += 256) {
    float s = row[j];
    s_all += expf(s - m_all);
    float sp = (lab[j] == li) ? s : -1e9f;
    s_pos += expf(sp - m_pos);
  }
  r1[t] = s_all; r2[t] = s_pos;
  __syncthreads();
  for (int off = 128; off > 0; off >>= 1) {
    if (t < off) { r1[t] += r1[t + off]; r2[t] += r2[t + off]; }
    __syncthreads();
  }
  if (t == 0)
    atomicAdd(acc, (logf(r1[0]) + m_all) - (logf(r2[0]) + m_pos));
}

__global__ void k_finalize(const float* buckets, const float* accs, float* loss,
                           int N, int T) {
  if (threadIdx.x == 0 && blockIdx.x == 0) {
    double ss = 0.0;
    for (int i = 0; i < 2048; ++i) ss += (double)buckets[i];
    double de = accs[0], ds = accs[1], nll = accs[2], cl = accs[3];
    double diag_loss = de / (double)N;
    double off_mean = (ss - ds) / ((double)N * (double)(N - 1));
    loss[0] = (float)(nll / (double)T + 0.1 * (diag_loss + off_mean) + cl / (double)T);
  }
}

// ---------------- host orchestration ----------------
extern "C" void kernel_launch(void* const* d_in, const int* in_sizes, int n_in,
                              void* d_out, int out_size, void* d_ws, size_t ws_size,
                              hipStream_t stream) {
  const int N = 16384, F = 512, H = 256, C = 16, E = 524288, T = 4096;
  const float TAU = 0.5f;

  const float* x        = (const float*)d_in[0];
  const int*   adj_row  = (const int*)d_in[1];
  const int*   adj_col  = (const int*)d_in[2];
  const float* adj_val  = (const float*)d_in[3];
  const int*   train_id = (const int*)d_in[4];
  const int*   y        = (const int*)d_in[5];
  const float* W_g1 = (const float*)d_in[6];  const float* b_g1 = (const float*)d_in[7];
  const float* W_l1 = (const float*)d_in[8];  const float* b_l1 = (const float*)d_in[9];
  const float* W_g2 = (const float*)d_in[10]; const float* b_g2 = (const float*)d_in[11];
  const float* W_l2 = (const float*)d_in[12]; const float* b_l2 = (const float*)d_in[13];
  const float* W_c1 = (const float*)d_in[14]; const float* b_c1 = (const float*)d_in[15];
  const float* W_c2 = (const float*)d_in[16]; const float* b_c2 = (const float*)d_in[17];

  float* z_out  = (float*)d_out;                  // [N,H]
  float* out_ls = z_out + (size_t)N * H;          // [N,C]
  float* loss   = out_ls + (size_t)N * C;         // [1]

  char* w = (char*)d_ws;
  auto carve = [&](size_t bytes) -> char* {
    char* p = w;
    w += (bytes + 255) & ~(size_t)255;
    return p;
  };

  unsigned short* x_b  = (unsigned short*)carve((size_t)N * F * 2);
  unsigned short* WgT1 = (unsigned short*)carve((size_t)H * F * 2);
  unsigned short* WgT2 = (unsigned short*)carve((size_t)H * F * 2);
  unsigned short* WlT1 = (unsigned short*)carve((size_t)H * H * 2);
  unsigned short* WlT2 = (unsigned short*)carve((size_t)H * H * 2);
  unsigned short* Wc1T = (unsigned short*)carve((size_t)H * H * 2);
  unsigned short* Wc2T = (unsigned short*)carve((size_t)C * H * 2);
  float* sup1 = (float*)carve((size_t)N * H * 4);
  float* sup2 = (float*)carve((size_t)N * H * 4);
  float* h1   = (float*)carve((size_t)N * H * 4);
  float* h2   = (float*)carve((size_t)N * H * 4);
  unsigned short* h1b = (unsigned short*)carve((size_t)N * H * 2);
  unsigned short* h2b = (unsigned short*)carve((size_t)N * H * 2);
  float* z1 = (float*)carve((size_t)N * H * 4);
  float* z2 = (float*)carve((size_t)N * H * 4);
  unsigned short* z_b  = (unsigned short*)carve((size_t)N * H * 2);
  unsigned short* z1nb = (unsigned short*)carve((size_t)N * H * 2);
  unsigned short* z2nb = (unsigned short*)carve((size_t)N * H * 2);
  float* zc = (float*)carve((size_t)N * H * 4);
  unsigned short* zcb = (unsigned short*)carve((size_t)N * H * 2);
  float* logits = (float*)carve((size_t)N * C * 4);
  unsigned short* At = (unsigned short*)carve((size_t)T * H * 2);
  unsigned short* Bt = (unsigned short*)carve((size_t)T * H * 2);
  int* lab_t = (int*)carve((size_t)T * 4);
  float* buckets = (float*)carve(2048 * 4);
  float* accs = (float*)carve(4 * 4);  // [diag_err, diag_sq, nll, cl]
  float* sim = (float*)carve((size_t)T * T * 4);

  const size_t nNH = (size_t)N * H;
  dim3 blk(256);
  auto gr = [](size_t n) { return dim3((unsigned)((n + 255) / 256)); };

  // zero accumulators / spmm outputs
  k_zero<<<gr(nNH), blk, 0, stream>>>(h1, nNH);
  k_zero<<<gr(nNH), blk, 0, stream>>>(h2, nNH);
  k_zero<<<gr(2048), blk, 0, stream>>>(buckets, 2048);
  k_zero<<<gr(4), blk, 0, stream>>>(accs, 4);

  // bf16 conversions / weight transposes
  k_cvt_bf16<<<gr((size_t)N * F), blk, 0, stream>>>(x, x_b, (size_t)N * F);
  k_wt<<<gr((size_t)F * H), blk, 0, stream>>>(W_g1, WgT1, F, H);
  k_wt<<<gr((size_t)F * H), blk, 0, stream>>>(W_g2, WgT2, F, H);
  k_wt<<<gr((size_t)H * H), blk, 0, stream>>>(W_l1, WlT1, H, H);
  k_wt<<<gr((size_t)H * H), blk, 0, stream>>>(W_l2, WlT2, H, H);
  k_wt<<<gr((size_t)H * H), blk, 0, stream>>>(W_c1, Wc1T, H, H);
  k_wt<<<gr((size_t)H * C), blk, 0, stream>>>(W_c2, Wc2T, H, C);

  // support = x @ W_g   (32x64 macro tiles: 512*4 = 2048 -> 256 blocks)
  k_gemm_nt24<<<256, blk, 0, stream>>>(x_b, WgT1, nullptr, sup1, N, H, F, 1.f);
  k_gemm_nt24<<<256, blk, 0, stream>>>(x_b, WgT2, nullptr, sup2, N, H, F, 1.f);

  // h = relu(spmm(support) + b_g)
  k_spmm<<<gr((size_t)E * 32), blk, 0, stream>>>(adj_val, adj_row, adj_col, sup1, h1, E, H);
  k_spmm<<<gr((size_t)E * 32), blk, 0, stream>>>(adj_val, adj_row, adj_col, sup2, h2, E, H);
  k_bias_relu<<<gr(nNH), blk, 0, stream>>>(h1, b_g1, nNH, H);
  k_bias_relu<<<gr(nNH), blk, 0, stream>>>(h2, b_g2, nNH, H);
  k_cvt_bf16<<<gr(nNH), blk, 0, stream>>>(h1, h1b, nNH);
  k_cvt_bf16<<<gr(nNH), blk, 0, stream>>>(h2, h2b, nNH);

  // z_k = h_k @ W_l_k + b_l_k
  k_gemm_nt24<<<256, blk, 0, stream>>>(h1b, WlT1, b_l1, z1, N, H, H, 1.f);
  k_gemm_nt24<<<256, blk, 0, stream>>>(h2b, WlT2, b_l2, z2, N, H, H, 1.f);

  // z = 0.5*(z1+z2) -> d_out + bf16 copy; row-normalize z1,z2 -> bf16
  k_avg<<<gr(nNH), blk, 0, stream>>>(z1, z2, z_out, z_b, nNH);
  k_rownorm<<<N / 8, blk, 0, stream>>>(z1, z1nb, N, H);
  k_rownorm<<<N / 8, blk, 0, stream>>>(z2, z2nb, N, H);

  // classifier: zc = z@W_c1+b_c1; logits = zc@W_c2+b_c2; log_softmax; nll
  k_gemm_nt24<<<256, blk, 0, stream>>>(z_b, Wc1T, b_c1, zc, N, H, H, 1.f);
  k_cvt_bf16<<<gr(nNH), blk, 0, stream>>>(zc, zcb, nNH);
  k_gemm_nt<<<128, blk, 0, stream>>>(zcb, Wc2T, b_c2, logits, N, C, H, 1.f);
  k_logsoftmax<<<gr(N), blk, 0, stream>>>(logits, out_ls, N, C);
  k_nll<<<gr(T), blk, 0, stream>>>(out_ls, train_id, y, accs + 2, T, C);

  // Barlow-Twins fused reduction: (256)^2 64x64 macro tiles / 8 = 8192 blocks
  k_bt_reduce44<<<8192, blk, 0, stream>>>(z1nb, z2nb, N, H, buckets, accs + 0, accs + 1);

  // SupCon: gather train rows, sim = (a@b^T)/tau, per-row lse reduction
  k_gather_rows<<<gr((size_t)T * H), blk, 0, stream>>>(z1nb, train_id, At, T, H);
  k_gather_rows<<<gr((size_t)T * H), blk, 0, stream>>>(z2nb, train_id, Bt, T, H);
  k_gather_lab<<<gr(T), blk, 0, stream>>>(y, train_id, lab_t, T);
  k_gemm_nt24<<<1024, blk, 0, stream>>>(At, Bt, nullptr, sim, T, T, H, 1.f / TAU);
  k_supcon<<<T, blk, 0, stream>>>(sim, lab_t, accs + 3, T);

  // combine: nll/T + 0.1*(diag_loss + off_mean) + cl/T
  k_finalize<<<1, 1, 0, stream>>>(buckets, accs, loss, N, T);

  (void)in_sizes; (void)n_in; (void)out_size; (void)ws_size;
}
